// AttentionBlock_65644280152459
// MI455X (gfx1250) — matrix operand order repaired and checked by reading, not compiled, not run on hardware
//
#include <hip/hip_runtime.h>
#include <hip/hip_bf16.h>

typedef __attribute__((ext_vector_type(16))) _Float16 v16h;
typedef __attribute__((ext_vector_type(8)))  float    v8f;

#define B_  4
#define H_  16
#define S_  2048
#define D_  64
#define KT  64            // key tile width
#define QT  64            // query tile per block (4 waves x 16 rows)
#define LSTR 68           // LDS row stride in halfs (64 + 4 pad)
#define MSTR 68           // LDS row stride in floats for mask bias

__global__ __launch_bounds__(128)
void AttentionBlock_65644280152459_kernel(
    const float* __restrict__ q,
    const float* __restrict__ k,
    const float* __restrict__ v,
    const float* __restrict__ att_mask,      // [S,S] broadcast over (b,h)
    const unsigned char* __restrict__ pmask, // [B,S,S] bool (1 byte)
    float* __restrict__ out)
{
    __shared__ _Float16 kbuf[KT * LSTR];               // [key][d]
    __shared__ _Float16 vbufT[D_ * LSTR];              // [d][key] (transposed V tile)
    __shared__ _Float16 pbuf[4 * 16 * LSTR];           // per-wave 16 x 64 P staging
    __shared__ __align__(16) float mbias[QT * MSTR];   // combined mask bias [q][k]

    const int tid  = threadIdx.x;
    const int lane = tid & 31;
    const int wave = tid >> 5;
    const int l16  = lane & 15;
    const int half = (lane >> 4) & 1;

    const int qtile = blockIdx.x;                 // 0..S/QT-1
    const int bh    = blockIdx.y;                 // 0..B*H-1
    const int b     = bh / H_;

    const int qbase = qtile * QT;
    const int wrow  = qbase + wave * 16;          // first query row of this wave

    const float scale = 0.125f;                   // 1/sqrt(64)

    const float* qp  = q + (size_t)bh * S_ * D_;
    const float* kp  = k + (size_t)bh * S_ * D_;
    const float* vp  = v + (size_t)bh * S_ * D_;
    const float* amp = att_mask + (size_t)qbase * S_;            // rows qbase..qbase+63
    const unsigned char* pmp = pmask + (size_t)b * S_ * S_ + (size_t)qbase * S_;

    // staging thread mapping: each thread covers rows {t*8 + srow}, cols [sc4, sc4+3]
    const int srow = tid >> 4;                    // 0..7
    const int sc4  = (tid & 15) << 2;             // 0..60

    // ---- Q fragments (A layout, 16x32 f16), scale folded in ----
    v16h a0, a1;
    {
        const float* qrowp = qp + (size_t)(wrow + l16) * D_;
        #pragma unroll
        for (int i = 0; i < 16; ++i) {
            int vr = i >> 1, j = i & 1;
            int kk = ((vr < 4) ? 0 : 16) + half * 8 + (vr & 3) * 2 + j;
            a0[i] = (_Float16)(qrowp[kk]      * scale);
            a1[i] = (_Float16)(qrowp[kk + 32] * scale);
        }
    }

    // constant all-ones B fragment: C += P . 1  gives softmax row sums in every lane
    v16h ones;
    #pragma unroll
    for (int i = 0; i < 16; ++i) ones[i] = (_Float16)1.0f;

    // online-softmax state: lane owns rows M = r + half*8 (r = 0..7), col = l16
    float mrow[8];
    v8f o[4], osum;
    v8f zacc = {};
    #pragma unroll
    for (int r = 0; r < 8; ++r) mrow[r] = -INFINITY;
    #pragma unroll
    for (int n = 0; n < 4; ++n) o[n] = zacc;
    osum = zacc;

    _Float16* pwave = pbuf + wave * 16 * LSTR;
    // bias in C layout: lane reads rows (wave*16 + half*8 + r), col l16 (+ n*16)
    const float* mbase = &mbias[(wave * 16 + half * 8) * MSTR + l16];

    for (int kt = 0; kt < S_ / KT; ++kt) {
        const int kbase = kt * KT;

        // single base pointers; per-t offsets are compile-time immediates
        const float* kb = kp + (size_t)kbase * D_ + srow * D_ + sc4;
        const float* vb = vp + (size_t)kbase * D_ + srow * D_ + sc4;
        const float* ab = amp + (size_t)srow * S_ + kbase + sc4;
        const unsigned char* pb = pmp + (size_t)srow * S_ + kbase + sc4;

        __syncthreads();
        // ---- stage K [key][d], V^T [d][key] (f16), mask bias (f32), coalesced ----
        #pragma unroll
        for (int t = 0; t < 8; ++t) {
            int row = t * 8 + srow;               // 0..63
            const float4 k4 = *(const float4*)(kb + t * (8 * D_));
            const float4 v4 = *(const float4*)(vb + t * (8 * D_));
            kbuf[row * LSTR + sc4 + 0] = (_Float16)k4.x;
            kbuf[row * LSTR + sc4 + 1] = (_Float16)k4.y;
            kbuf[row * LSTR + sc4 + 2] = (_Float16)k4.z;
            kbuf[row * LSTR + sc4 + 3] = (_Float16)k4.w;
            vbufT[(sc4 + 0) * LSTR + row] = (_Float16)v4.x;
            vbufT[(sc4 + 1) * LSTR + row] = (_Float16)v4.y;
            vbufT[(sc4 + 2) * LSTR + row] = (_Float16)v4.z;
            vbufT[(sc4 + 3) * LSTR + row] = (_Float16)v4.w;
            // combined mask bias: pad ? -inf : att_mask
            const float4  m4 = *(const float4*)(ab + t * (8 * S_));
            unsigned int  p4 = *(const unsigned int*)(pb + t * (8 * S_));
            float4 bias;
            bias.x = (p4 & 0x000000FFu) ? -INFINITY : m4.x;
            bias.y = (p4 & 0x0000FF00u) ? -INFINITY : m4.y;
            bias.z = (p4 & 0x00FF0000u) ? -INFINITY : m4.z;
            bias.w = (p4 & 0xFF000000u) ? -INFINITY : m4.w;
            *(float4*)(&mbias[row * MSTR + sc4]) = bias;
        }
        __syncthreads();

        // prefetch next K/V/mask tiles into cache hierarchy
        if (kt + 1 < S_ / KT) {
            __builtin_prefetch(kb + KT * D_ + (tid & 3) * 8, 0, 1);
            __builtin_prefetch(vb + KT * D_ + (tid & 3) * 8, 0, 1);
            __builtin_prefetch(ab + KT + (tid & 3) * 16, 0, 1);
        }

        // ---- S = (Q*scale) . K^T + bias : bias preloaded as the WMMA C operand ----
        v8f sacc[4];
        #pragma unroll
        for (int n = 0; n < 4; ++n) {
            // C-layout bias load (merges to ds_load_2addr_b32 pairs)
            v8f c;
            #pragma unroll
            for (int r = 0; r < 8; ++r)
                c[r] = mbase[r * MSTR + n * 16];
            v16h b0, b1;
            const _Float16* krow = &kbuf[(n * 16 + l16) * LSTR];
            #pragma unroll
            for (int i = 0; i < 16; ++i) {        // B layout: K = (i&8)*2 + half*8 + (i&7)
                int dd = (i & 8) * 2 + half * 8 + (i & 7);
                b0[i] = krow[dd];
                b1[i] = krow[dd + 32];
            }
            c = __builtin_amdgcn_wmma_f32_16x16x32_f16(false, a0, false, b0, (short)0, c, false, false);
            c = __builtin_amdgcn_wmma_f32_16x16x32_f16(false, a1, false, b1, (short)0, c, false, false);
            sacc[n] = c;
        }

        // ---- online softmax (per row, short-lived temps) ----
        #pragma unroll
        for (int r = 0; r < 8; ++r) {
            float x0 = sacc[0][r];
            float x1 = sacc[1][r];
            float x2 = sacc[2][r];
            float x3 = sacc[3][r];
            float tmax = fmaxf(fmaxf(x0, x1), fmaxf(x2, x3));
            tmax = fmaxf(tmax, __shfl_xor(tmax, 1));
            tmax = fmaxf(tmax, __shfl_xor(tmax, 2));
            tmax = fmaxf(tmax, __shfl_xor(tmax, 4));
            tmax = fmaxf(tmax, __shfl_xor(tmax, 8));
            float newm   = fmaxf(mrow[r], tmax);
            float newm_s = fmaxf(newm, -3.0e38f);   // finite shift: exp(-inf - newm_s) == 0
            float corr   = __expf(fmaxf(mrow[r], -3.0e38f) - newm_s);
            mrow[r] = newm;
            x0 = __expf(x0 - newm_s);
            x1 = __expf(x1 - newm_s);
            x2 = __expf(x2 - newm_s);
            x3 = __expf(x3 - newm_s);
            _Float16* pw = &pwave[(r + half * 8) * LSTR + l16];
            pw[0]  = (_Float16)x0;
            pw[16] = (_Float16)x1;
            pw[32] = (_Float16)x2;
            pw[48] = (_Float16)x3;
            o[0][r] *= corr;
            o[1][r] *= corr;
            o[2][r] *= corr;
            o[3][r] *= corr;
            osum[r] *= corr;
        }

        asm volatile("s_wait_dscnt 0" ::: "memory");

        // ---- read P back as A-layout fragments (wave-local) ----
        v16h pa0, pa1;
        {
            const _Float16* prow = &pwave[l16 * LSTR];
            #pragma unroll
            for (int i = 0; i < 16; ++i) {
                int vr = i >> 1, j = i & 1;
                int kk = ((vr < 4) ? 0 : 16) + half * 8 + (vr & 3) * 2 + j;
                pa0[i] = prow[kk];
                pa1[i] = prow[kk + 32];
            }
        }

        // ---- O += P . V (8 WMMA)  and  osum += P . 1 (2 WMMA, no LDS reads) ----
        #pragma unroll
        for (int n = 0; n < 4; ++n) {
            v16h bv0, bv1;
            const _Float16* vcol = &vbufT[(n * 16 + l16) * LSTR];
            #pragma unroll
            for (int i = 0; i < 16; ++i) {        // B layout: K = (i&8)*2 + half*8 + (i&7)
                int kk = (i & 8) * 2 + half * 8 + (i & 7);
                bv0[i] = vcol[kk];
                bv1[i] = vcol[kk + 32];
            }
            o[n] = __builtin_amdgcn_wmma_f32_16x16x32_f16(false, pa0, false, bv0, (short)0, o[n], false, false);
            o[n] = __builtin_amdgcn_wmma_f32_16x16x32_f16(false, pa1, false, bv1, (short)0, o[n], false, false);
        }
        osum = __builtin_amdgcn_wmma_f32_16x16x32_f16(false, pa0, false, ones, (short)0, osum, false, false);
        osum = __builtin_amdgcn_wmma_f32_16x16x32_f16(false, pa1, false, ones, (short)0, osum, false, false);
    }

    // ---- normalize and store (fp32 out); every lane holds its row's sum in osum[r] ----
    float* op = out + (size_t)bh * S_ * D_;
    #pragma unroll
    for (int r = 0; r < 8; ++r) {
        float inv  = 1.0f / osum[r];
        int   qrow = wrow + r + half * 8;
        #pragma unroll
        for (int n = 0; n < 4; ++n)
            op[(size_t)qrow * D_ + n * 16 + l16] = o[n][r] * inv;
    }
}

extern "C" void kernel_launch(void* const* d_in, const int* in_sizes, int n_in,
                              void* d_out, int out_size, void* d_ws, size_t ws_size,
                              hipStream_t stream) {
    (void)in_sizes; (void)n_in; (void)d_ws; (void)ws_size; (void)out_size;
    const float*         q  = (const float*)d_in[0];
    const float*         k  = (const float*)d_in[1];
    const float*         v  = (const float*)d_in[2];
    const float*         am = (const float*)d_in[3];
    const unsigned char* pm = (const unsigned char*)d_in[4];  // jnp bool -> 1 byte
    float* out = (float*)d_out;

    dim3 grid(S_ / QT, B_ * H_);   // 32 x 64 blocks
    dim3 block(128);               // 4 waves (wave32)
    AttentionBlock_65644280152459_kernel<<<grid, block, 0, stream>>>(q, k, v, am, pm, out);
}
